// CharRNN_85624468013090
// MI455X (gfx1250) — matrix operand order repaired and checked by reading, not compiled, not run on hardware
//
#include <hip/hip_runtime.h>
#include <math.h>

// Problem dims (fixed by reference)
#define B_   64
#define T_   512
#define I_   128
#define H_   1024
#define G4H  (4 * H_)

#define STEP_BLOCKS 32   // persistent kernel: 32 WGs x 8 waves = 256 waves

typedef __attribute__((ext_vector_type(16))) __bf16 bf16x16;
typedef __attribute__((ext_vector_type(8)))  float  floatx8;

// ---------- helpers ----------

__device__ __forceinline__ unsigned short f2bf(float f) {
    // round-to-nearest-even f32 -> bf16
    unsigned int u = __float_as_uint(f);
    u += 0x7FFFu + ((u >> 16) & 1u);
    return (unsigned short)(u >> 16);
}

__device__ __forceinline__ bf16x16 load_frag(const unsigned short* lo,
                                             const unsigned short* hi) {
    union { uint4 u[2]; bf16x16 v; } f;
    f.u[0] = *reinterpret_cast<const uint4*>(lo);   // 8 bf16
    f.u[1] = *reinterpret_cast<const uint4*>(hi);   // 8 bf16
    return f.v;
}

__device__ __forceinline__ float sigmoidf_(float x) {
    return 1.0f / (1.0f + __expf(-x));
}

// Device-scope grid barrier: monotonic ticket counter, all 32 WGs resident.
__device__ __forceinline__ void grid_sync(unsigned int* cnt, unsigned int target) {
    __syncthreads();
    __threadfence();   // release: make this step's h/hs stores visible device-wide
    if (threadIdx.x == 0) {
        __hip_atomic_fetch_add(cnt, 1u, __ATOMIC_RELEASE, __HIP_MEMORY_SCOPE_AGENT);
        while (__hip_atomic_load(cnt, __ATOMIC_ACQUIRE, __HIP_MEMORY_SCOPE_AGENT) < target) {
            __builtin_amdgcn_s_sleep(2);
        }
    }
    __syncthreads();
    __threadfence();   // acquire: drop stale lines before reading peers' h
}

// ---------- prep kernels ----------

__global__ __launch_bounds__(256) void cvt_f32_bf16_kernel(
        const float* __restrict__ src, unsigned short* __restrict__ dst, int n) {
    int idx = blockIdx.x * blockDim.x + threadIdx.x;
    if (idx < n) dst[idx] = f2bf(src[idx]);
}

__global__ __launch_bounds__(256) void prep_state_kernel(
        const float* __restrict__ b_ih, const float* __restrict__ b_hh,
        float* __restrict__ bsum,
        unsigned short* __restrict__ h0, unsigned short* __restrict__ h1,
        unsigned int* __restrict__ barrier_cnt) {
    int idx = blockIdx.x * blockDim.x + threadIdx.x;
    if (idx == 0) *barrier_cnt = 0u;
    if (idx < G4H) bsum[idx] = b_ih[idx] + b_hh[idx];
    if (idx < B_ * H_) {
        h0[idx] = 0;
        h1[idx] = 0;
    }
}

// ---------- persistent LSTM scan ----------
// One launch runs all T=512 steps. Per step:
//   gates[64,4096] = x_t @ W_ih^T + h_in @ W_hh^T + bsum, then cell update.
// One wave owns a 16(batch) x 16(h-col) tile and computes all 4 gate tiles
// (i,f,g,o); in the C/D layout all four accumulators align element-wise per
// lane, so the cell is pure per-lane VALU and the c-state stays in VGPRs for
// the whole sequence. h double-buffers through global memory with one
// device-scope barrier per step. 32 WGs -> always co-resident, no deadlock.

__global__ __launch_bounds__(256) void lstm_scan_kernel(
        const unsigned short* __restrict__ x_bf,   // [B,T,I] bf16
        const unsigned short* __restrict__ Wih,    // [4H,I]  bf16
        const unsigned short* __restrict__ Whh,    // [4H,H]  bf16
        const float*          __restrict__ bsum,   // [4H]
        unsigned short*       __restrict__ h0,     // [B,H] bf16 (even-t input)
        unsigned short*       __restrict__ h1,     // [B,H] bf16 (odd-t input)
        unsigned short*       __restrict__ hs,     // [B,T,H] bf16
        unsigned int*         __restrict__ barrier_cnt) {
    const int lane   = threadIdx.x & 31;
    const int w      = blockIdx.x * 8 + (threadIdx.x >> 5);
    const int m0     = (w & 3) * 16;      // batch tile base
    const int n0     = (w >> 2) * 16;     // h-column tile base
    const int l15    = lane & 15;
    const int hiHalf = lane >> 4;         // 0: lanes 0-15, 1: lanes 16-31
    const int akoff  = hiHalf * 8;        // A-frag K offset of low half
    const int bkoff  = hiHalf * 8;        // B-frag K offset
    const int arow   = m0 + l15;          // batch row this lane loads for A
    const int brow   = n0 + l15;          // gate-row this lane loads for B

    // Loop-invariant bases (set up once for all 512 steps).
    const unsigned short* xbase = x_bf + (size_t)arow * T_ * I_;
    const unsigned short* wih_b[4];
    const unsigned short* whh_b[4];
#pragma unroll
    for (int q = 0; q < 4; ++q) {
        wih_b[q] = Wih + (size_t)(q * H_ + brow) * I_ + bkoff;
        whh_b[q] = Whh + (size_t)(q * H_ + brow) * H_ + bkoff;
    }
    const int   hcol = n0 + l15;
    const float bi = bsum[0 * H_ + hcol];
    const float bf = bsum[1 * H_ + hcol];
    const float bg = bsum[2 * H_ + hcol];
    const float bo = bsum[3 * H_ + hcol];

    // c-state lives in registers for the entire sequence.
    float c_reg[8];
#pragma unroll
    for (int r = 0; r < 8; ++r) c_reg[r] = 0.0f;

    for (int t = 0; t < T_; ++t) {
        const unsigned short* h_in  = (t & 1) ? h1 : h0;
        unsigned short*       h_out = (t & 1) ? h0 : h1;

        floatx8 acc[4] = {};              // i, f, g, o accumulators (f32)

        // ---- x_t contribution: K = I_ = 128 ----
        const unsigned short* arowp = xbase + (size_t)t * I_;
        for (int k0 = 0; k0 < I_; k0 += 32) {
            bf16x16 a = load_frag(arowp + k0 + akoff, arowp + k0 + akoff + 16);
#pragma unroll
            for (int q = 0; q < 4; ++q) {
                const unsigned short* bp = wih_b[q] + k0;
                bf16x16 b = load_frag(bp, bp + 16);
                acc[q] = __builtin_amdgcn_wmma_f32_16x16x32_bf16(
                    false, a, false, b, (short)0, acc[q], false, false);
            }
        }

        // ---- h contribution: K = H_ = 1024 ----
        const unsigned short* hrowp = h_in + (size_t)arow * H_;
        for (int k0 = 0; k0 < H_; k0 += 32) {
            bf16x16 a = load_frag(hrowp + k0 + akoff, hrowp + k0 + akoff + 16);
#pragma unroll
            for (int q = 0; q < 4; ++q) {
                const unsigned short* bp = whh_b[q] + k0;
                bf16x16 b = load_frag(bp, bp + 16);
                acc[q] = __builtin_amdgcn_wmma_f32_16x16x32_bf16(
                    false, a, false, b, (short)0, acc[q], false, false);
            }
        }

        // ---- bias + LSTM cell (element-wise per lane in C/D layout) ----
#pragma unroll
        for (int r = 0; r < 8; ++r) {
            const int m = m0 + r + hiHalf * 8;   // batch index of this element
            float iv = sigmoidf_(acc[0][r] + bi);
            float fv = sigmoidf_(acc[1][r] + bf);
            float gv = tanhf(acc[2][r] + bg);
            float ov = sigmoidf_(acc[3][r] + bo);
            float cn = fv * c_reg[r] + iv * gv;
            c_reg[r] = cn;
            unsigned short hb = f2bf(ov * tanhf(cn));
            h_out[(size_t)m * H_ + hcol] = hb;
            hs[((size_t)m * T_ + t) * H_ + hcol] = hb;
        }

        // Separate write h(t) from read h(t) at step t+1 (buffers swap).
        grid_sync(barrier_cnt, (unsigned int)(STEP_BLOCKS * (t + 1)));
    }
}

// ---------- final FC: out[B*T, I] = hs @ fc_w^T + fc_b ----------
// Grid: 2048 blocks x 256 threads = 16384 waves = 2048 m-tiles x 8 n-tiles.

__global__ __launch_bounds__(256) void fc_kernel(
        const unsigned short* __restrict__ hs,    // [B*T, H] bf16
        const unsigned short* __restrict__ Wfc,   // [I, H]   bf16
        const float*          __restrict__ fcb,   // [I]
        float*                __restrict__ out) { // [B*T, I]
    const int lane   = threadIdx.x & 31;
    const int w      = blockIdx.x * 8 + (threadIdx.x >> 5);
    const int n0     = (w & 7) * 16;
    const int m0     = (w >> 3) * 16;
    const int l15    = lane & 15;
    const int hiHalf = lane >> 4;
    const int akoff  = hiHalf * 8;
    const int bkoff  = hiHalf * 8;

    const unsigned short* arowp = hs + (size_t)(m0 + l15) * H_;
    const unsigned short* browp = Wfc + (size_t)(n0 + l15) * H_;

    floatx8 acc = {};
    for (int k0 = 0; k0 < H_; k0 += 32) {
        bf16x16 a = load_frag(arowp + k0 + akoff, arowp + k0 + akoff + 16);
        const unsigned short* bp = browp + k0 + bkoff;
        bf16x16 b = load_frag(bp, bp + 16);
        acc = __builtin_amdgcn_wmma_f32_16x16x32_bf16(
            false, a, false, b, (short)0, acc, false, false);
    }

    const float bias = fcb[n0 + l15];
#pragma unroll
    for (int r = 0; r < 8; ++r) {
        const int m = m0 + r + hiHalf * 8;
        out[(size_t)m * I_ + n0 + l15] = acc[r] + bias;
    }
}

// ---------- launch ----------

extern "C" void kernel_launch(void* const* d_in, const int* in_sizes, int n_in,
                              void* d_out, int out_size, void* d_ws, size_t ws_size,
                              hipStream_t stream) {
    const float* x    = (const float*)d_in[0];  // [B,T,I]
    const float* W_ih = (const float*)d_in[1];  // [4H,I]
    const float* W_hh = (const float*)d_in[2];  // [4H,H]
    const float* b_ih = (const float*)d_in[3];  // [4H]
    const float* b_hh = (const float*)d_in[4];  // [4H]
    const float* fc_w = (const float*)d_in[5];  // [I,H]
    const float* fc_b = (const float*)d_in[6];  // [I]
    float* out = (float*)d_out;

    // Workspace carve-up (256B aligned), ~85 MB total.
    char*  ws  = (char*)d_ws;
    size_t off = 0;
    auto alloc = [&](size_t bytes) {
        size_t o = off;
        off = (off + bytes + 255) & ~(size_t)255;
        return o;
    };
    unsigned short* x_bf   = (unsigned short*)(ws + alloc((size_t)B_ * T_ * I_ * 2));
    unsigned short* Wih_bf = (unsigned short*)(ws + alloc((size_t)G4H * I_ * 2));
    unsigned short* Whh_bf = (unsigned short*)(ws + alloc((size_t)G4H * H_ * 2));
    unsigned short* Wfc_bf = (unsigned short*)(ws + alloc((size_t)I_ * H_ * 2));
    float*          bsum   = (float*)         (ws + alloc((size_t)G4H * 4));
    unsigned short* h0     = (unsigned short*)(ws + alloc((size_t)B_ * H_ * 2));
    unsigned short* h1     = (unsigned short*)(ws + alloc((size_t)B_ * H_ * 2));
    unsigned int*   bcnt   = (unsigned int*)  (ws + alloc(256));
    unsigned short* hs     = (unsigned short*)(ws + alloc((size_t)B_ * T_ * H_ * 2));
    (void)ws_size; (void)in_sizes; (void)n_in; (void)out_size;

    // Down-convert operands to bf16 (fresh every call: deterministic).
    auto launch_cvt = [&](const float* s, unsigned short* d, int n) {
        cvt_f32_bf16_kernel<<<(n + 255) / 256, 256, 0, stream>>>(s, d, n);
    };
    launch_cvt(x,    x_bf,   B_ * T_ * I_);
    launch_cvt(W_ih, Wih_bf, G4H * I_);
    launch_cvt(W_hh, Whh_bf, G4H * H_);
    launch_cvt(fc_w, Wfc_bf, I_ * H_);
    prep_state_kernel<<<(B_ * H_ + 255) / 256, 256, 0, stream>>>(
        b_ih, b_hh, bsum, h0, h1, bcnt);

    // Entire 512-step scan in ONE persistent launch; the sequential
    // dependency is carried by an in-kernel device-scope barrier instead of
    // 512 kernel-launch latencies.
    lstm_scan_kernel<<<STEP_BLOCKS, 256, 0, stream>>>(
        x_bf, Wih_bf, Whh_bf, bsum, h0, h1, hs, bcnt);

    // Final projection back to I=128.
    fc_kernel<<<(B_ * T_ / 16) * (I_ / 16) / 8, 256, 0, stream>>>(
        hs, Wfc_bf, fc_b, out);
}
